// Mamba2_13752485282233
// MI455X (gfx1250) — matrix operand order repaired and checked by reading, not compiled, not run on hardware
//
#include <hip/hip_runtime.h>
#include <hip/hip_bf16.h>

#define S_LEN   2048
#define B_SZ    2
#define HID     1024
#define NH      32
#define HP      64
#define INTER   2048
#define NSTATE  128
#define KCONV   4
#define CONV_DIM 2304
#define PROJ    4384
#define PROJ_PAD 4416           /* PROJ rounded up to 64 */
#define QCH     64
#define NCH     (S_LEN / QCH)   /* 32 chunks */
#define ROWS    (B_SZ * S_LEN)  /* 4096 */

typedef __attribute__((ext_vector_type(16))) __bf16 v16bf;
typedef __attribute__((ext_vector_type(8)))  __bf16 v8bf;
typedef __attribute__((ext_vector_type(8)))  float  v8f;

__device__ __forceinline__ v8f wmma_bf16(v16bf a, v16bf b, v8f c) {
  return __builtin_amdgcn_wmma_f32_16x16x32_bf16(false, a, false, b, (short)0, c, false, false);
}

__device__ __forceinline__ unsigned lds_off(const void* p) {
  return (unsigned)(uintptr_t)p;   // LDS aperture: addr[31:0] is the LDS byte offset
}

// async global->LDS copy of 16 bytes per lane (tracked by ASYNCcnt)
__device__ __forceinline__ void async_g2l_b128(unsigned ldsaddr, const void* gaddr) {
  asm volatile("global_load_async_to_lds_b128 %0, %1, off"
               :: "v"(ldsaddr), "v"(gaddr) : "memory");
}
__device__ __forceinline__ void wait_async0() {
  asm volatile("s_wait_asynccnt 0" ::: "memory");
}

__device__ __forceinline__ v16bf frag_from(const __bf16* p0, const __bf16* p1) {
  union { v16bf v; v8bf h[2]; } u;
  u.h[0] = *(const v8bf*)p0;
  u.h[1] = *(const v8bf*)p1;
  return u.v;
}

// A fragment 16x32: element (m,k) at base[m*ld + k]; ld multiple of 8.
__device__ __forceinline__ v16bf load_frag_a(const __bf16* base, int ld, int m0, int k0, int lane) {
  const int r  = m0 + (lane & 15);
  const int kb = k0 + ((lane >> 4) << 3);
  const __bf16* p = base + r * ld + kb;
  return frag_from(p, p + 16);
}

// B fragment 32x16: element (k,n) at base[n*ld + k]; ld multiple of 8.
__device__ __forceinline__ v16bf load_frag_bt(const __bf16* base, int ld, int k0, int n0, int lane) {
  const int c  = n0 + (lane & 15);
  const int kb = k0 + ((lane >> 4) << 3);
  const __bf16* p = base + c * ld + kb;
  return frag_from(p, p + 16);
}

// ----------------------------------------------------------------------------
// f32 -> bf16 conversion with optional zero padding
// ----------------------------------------------------------------------------
__global__ void f32_to_bf16_kernel(const float* __restrict__ in, __bf16* __restrict__ out,
                                   long nsrc, long ntot) {
  long i = (long)blockIdx.x * 256 + threadIdx.x;
  if (i >= ntot) return;
  out[i] = (i < nsrc) ? (__bf16)in[i] : (__bf16)0.0f;
}

// ----------------------------------------------------------------------------
// GEMM: Out[M,N] = A[M,K] @ Bn[N,K]^T + bias   (bf16 inputs, async-LDS staging)
// Bn has >= ceil(N/64)*64 rows (padding zero-filled by caller).
// ----------------------------------------------------------------------------
__global__ void __launch_bounds__(256)
gemm_bf16(const __bf16* __restrict__ Abf, const __bf16* __restrict__ Bn,
          const float* __restrict__ bias, float* __restrict__ Out,
          int M, int Nn, int Kk) {
  __shared__ alignas(16) __bf16 As[64 * 40];  // [m][k]
  __shared__ alignas(16) __bf16 Bt[64 * 40];  // [n][k]
  const int tid = threadIdx.x, lane = tid & 31, w = tid >> 5;
  const int m0 = blockIdx.x * 64, n0 = blockIdx.y * 64;
  const int tr = w & 3;
  const int tc = (w >> 2) * 2;
  v8f acc0 = {}, acc1 = {};
  const int sr = tid >> 2, skb = (tid & 3) * 8;   // staging: 16B per thread per tile
  for (int k0 = 0; k0 < Kk; k0 += 32) {
    async_g2l_b128(lds_off(As + sr * 40 + skb),
                   Abf + (size_t)(m0 + sr) * Kk + k0 + skb);
    async_g2l_b128(lds_off(Bt + sr * 40 + skb),
                   Bn + (size_t)(n0 + sr) * Kk + k0 + skb);
    wait_async0();
    __syncthreads();
    v16bf a  = load_frag_a(As, 40, tr * 16, 0, lane);
    v16bf b0 = load_frag_bt(Bt, 40, 0, tc * 16, lane);
    v16bf b1 = load_frag_bt(Bt, 40, 0, tc * 16 + 16, lane);
    acc0 = wmma_bf16(a, b0, acc0);
    acc1 = wmma_bf16(a, b1, acc1);
    __syncthreads();
  }
  const int hi = lane >> 4, col = lane & 15;
#pragma unroll
  for (int i = 0; i < 8; ++i) {
    int m  = m0 + tr * 16 + i + 8 * hi;
    int n1 = n0 + tc * 16 + col;
    int n2 = n1 + 16;
    if (n1 < Nn) Out[(size_t)m * Nn + n1] = acc0[i] + bias[n1];
    if (n2 < Nn) Out[(size_t)m * Nn + n2] = acc1[i] + bias[n2];
  }
}

// ----------------------------------------------------------------------------
// depthwise causal conv (K=4) + SiLU
// ----------------------------------------------------------------------------
__global__ void conv_silu_kernel(const float* __restrict__ proj, const float* __restrict__ cw,
                                 float* __restrict__ hBC) {
  long idx = (long)blockIdx.x * 256 + threadIdx.x;
  if (idx >= (long)ROWS * CONV_DIM) return;
  int  cdim = (int)(idx % CONV_DIM);
  long row  = idx / CONV_DIM;
  int  b = (int)(row / S_LEN), s = (int)(row % S_LEN);
  float acc = 0.f;
#pragma unroll
  for (int i = 0; i < KCONV; ++i) {
    int ss = s - (KCONV - 1) + i;
    if (ss >= 0)
      acc += proj[(size_t)(b * S_LEN + ss) * PROJ + INTER + cdim] * cw[cdim * KCONV + i];
  }
  hBC[idx] = acc / (1.0f + expf(-acc));
}

__global__ void dt_softplus_kernel(const float* __restrict__ proj, const float* __restrict__ dt_bias,
                                   float* __restrict__ dts) {
  int idx = blockIdx.x * 256 + threadIdx.x;
  if (idx >= ROWS * NH) return;
  int h = idx & (NH - 1), row = idx >> 5;
  float x = proj[(size_t)row * PROJ + (INTER + CONV_DIM) + h] + dt_bias[h];
  dts[idx] = (x > 20.f) ? x : log1pf(expf(x));
}

// inclusive scan of QCH floats in LDS
__device__ __forceinline__ void scan64(float* s, int tid) {
#pragma unroll
  for (int off = 1; off < QCH; off <<= 1) {
    float v = 0.f;
    if (tid >= off && tid < QCH) v = s[tid - off];
    __syncthreads();
    if (tid >= off && tid < QCH) s[tid] += v;
    __syncthreads();
  }
}

// ----------------------------------------------------------------------------
// pass1: per (b,chunk,head): local_state[p][n] = sum_l Bdec[l,n] * (h*dt)[l,p]
// ----------------------------------------------------------------------------
__global__ void __launch_bounds__(256)
ssd_pass1(const float* __restrict__ hBC, const float* __restrict__ dts,
          const float* __restrict__ A_log, __bf16* __restrict__ locS,
          float* __restrict__ Alast) {
  __shared__ float sAcum[QCH];
  __shared__ alignas(16) __bf16 sB[NSTATE * 72]; // Bdec  [n][l]
  __shared__ alignas(16) __bf16 sH[HP * 72];     // h*dt  [p][l]
  const int tid = threadIdx.x, lane = tid & 31, w = tid >> 5;
  const int h = blockIdx.x % NH;
  const int c = (blockIdx.x / NH) % NCH;
  const int b = blockIdx.x / (NH * NCH);
  const int t0 = c * QCH;
  const float Aneg = -expf(A_log[h]);
  if (tid < QCH) sAcum[tid] = Aneg * dts[(size_t)(b * S_LEN + t0 + tid) * NH + h];
  __syncthreads();
  scan64(sAcum, tid);
  const float alast = sAcum[QCH - 1];
  for (int i = tid; i < NSTATE * (QCH / 8); i += 256) {
    int n = i >> 3, lg = (i & 7) * 8;
    v8bf v;
#pragma unroll
    for (int e = 0; e < 8; ++e) {
      int l = lg + e;
      float bv = hBC[(size_t)(b * S_LEN + t0 + l) * CONV_DIM + INTER + n];
      v[e] = (__bf16)(bv * expf(alast - sAcum[l]));
    }
    *(v8bf*)(sB + n * 72 + lg) = v;
  }
  for (int i = tid; i < HP * (QCH / 8); i += 256) {
    int p = i >> 3, lg = (i & 7) * 8;
    v8bf v;
#pragma unroll
    for (int e = 0; e < 8; ++e) {
      int l = lg + e;
      size_t t = (size_t)(b * S_LEN + t0 + l);
      v[e] = (__bf16)(hBC[t * CONV_DIM + h * HP + p] * dts[t * NH + h]);
    }
    *(v8bf*)(sH + p * 72 + lg) = v;
  }
  __syncthreads();
  v8f acc[4] = {};
  for (int l0 = 0; l0 < QCH; l0 += 32) {
    v16bf a = load_frag_a(sB, 72, w * 16, l0, lane);   // rows = n, k = l
#pragma unroll
    for (int pt = 0; pt < 4; ++pt) {
      v16bf bb = load_frag_bt(sH, 72, l0, pt * 16, lane);
      acc[pt] = wmma_bf16(a, bb, acc[pt]);
    }
  }
  const int hi = lane >> 4, col = lane & 15;
  size_t base = ((size_t)(b * NCH + c) * NH + h) * (NSTATE * HP);
#pragma unroll
  for (int pt = 0; pt < 4; ++pt)
#pragma unroll
    for (int i = 0; i < 8; ++i) {
      int n = w * 16 + i + 8 * hi;
      int p = pt * 16 + col;
      locS[base + (size_t)p * NSTATE + n] = (__bf16)acc[pt][i];   // [p][n] layout
    }
  if (tid == 0) Alast[(b * NCH + c) * NH + h] = alast;
}

// ----------------------------------------------------------------------------
// pass2: sequential cross-chunk recurrence per (b,h)
// ----------------------------------------------------------------------------
__global__ void __launch_bounds__(256)
ssd_pass2(const __bf16* __restrict__ locS, const float* __restrict__ Alast,
          __bf16* __restrict__ sIn) {
  const int b = blockIdx.x >> 5, h = blockIdx.x & 31;
  const int tid = threadIdx.x;
  float run[32];
#pragma unroll
  for (int j = 0; j < 32; ++j) run[j] = 0.f;
  for (int c = 0; c < NCH; ++c) {
    size_t base = ((size_t)(b * NCH + c) * NH + h) * (NSTATE * HP) + (size_t)tid * 32;
    v8bf* dst = (v8bf*)(sIn + base);
    const v8bf* src = (const v8bf*)(locS + base);
#pragma unroll
    for (int q = 0; q < 4; ++q) {
      v8bf o;
#pragma unroll
      for (int e = 0; e < 8; ++e) o[e] = (__bf16)run[q * 8 + e];
      dst[q] = o;
    }
    float dec = expf(Alast[(b * NCH + c) * NH + h]);
#pragma unroll
    for (int q = 0; q < 4; ++q) {
      v8bf lv = src[q];
#pragma unroll
      for (int e = 0; e < 8; ++e) run[q * 8 + e] = run[q * 8 + e] * dec + (float)lv[e];
    }
  }
}

// ----------------------------------------------------------------------------
// pass3: per (b,chunk,head): Y = M@(h*dt) + exp(Acum)*(C@stateIn) + D*h0
// ----------------------------------------------------------------------------
__global__ void __launch_bounds__(256)
ssd_pass3(const float* __restrict__ hBC, const float* __restrict__ dts,
          const float* __restrict__ A_log, const __bf16* __restrict__ sInG,
          const float* __restrict__ Dp, float* __restrict__ yb) {
  __shared__ float sAcum[QCH];
  __shared__ alignas(16) __bf16 sC[QCH * 136];    // C [l][n]
  __shared__ alignas(16) __bf16 sBS[QCH * 136];   // B [l][n], later stateIn [p][n]
  __shared__ alignas(16) __bf16 sH[HP * 72];      // h*dt [p][l]
  __shared__ alignas(16) __bf16 sM[QCH * 72];     // masked decay matrix [l][s]
  const int tid = threadIdx.x, lane = tid & 31, w = tid >> 5;
  const int h = blockIdx.x % NH;
  const int c = (blockIdx.x / NH) % NCH;
  const int b = blockIdx.x / (NH * NCH);
  const int t0 = c * QCH;
  const float Aneg = -expf(A_log[h]);
  if (tid < QCH) sAcum[tid] = Aneg * dts[(size_t)(b * S_LEN + t0 + tid) * NH + h];
  __syncthreads();
  scan64(sAcum, tid);
  for (int i = tid; i < QCH * (NSTATE / 8); i += 256) {
    int l = i >> 4, ng = (i & 15) * 8;
    size_t t = (size_t)(b * S_LEN + t0 + l);
    const float* src = hBC + t * CONV_DIM + INTER;
    v8bf vc, vb;
#pragma unroll
    for (int e = 0; e < 8; ++e) {
      vb[e] = (__bf16)src[ng + e];
      vc[e] = (__bf16)src[NSTATE + ng + e];
    }
    *(v8bf*)(sC + l * 136 + ng)  = vc;
    *(v8bf*)(sBS + l * 136 + ng) = vb;
  }
  for (int i = tid; i < HP * (QCH / 8); i += 256) {
    int p = i >> 3, lg = (i & 7) * 8;
    v8bf v;
#pragma unroll
    for (int e = 0; e < 8; ++e) {
      int l = lg + e;
      size_t t = (size_t)(b * S_LEN + t0 + l);
      v[e] = (__bf16)(hBC[t * CONV_DIM + h * HP + p] * dts[t * NH + h]);
    }
    *(v8bf*)(sH + p * 72 + lg) = v;
  }
  __syncthreads();
  const int hi = lane >> 4, col = lane & 15;
  // CB phase: CB[l,s] = sum_n C[l,n]*B[s,n]; mask + decay -> sM[l][s]
#pragma unroll
  for (int q = 0; q < 2; ++q) {
    int ti = w * 2 + q, lt = ti >> 2, st = ti & 3;
    v8f acc = {};
#pragma unroll
    for (int n0 = 0; n0 < NSTATE; n0 += 32) {
      v16bf a  = load_frag_a(sC, 136, lt * 16, n0, lane);
      v16bf bb = load_frag_bt(sBS, 136, n0, st * 16, lane);
      acc = wmma_bf16(a, bb, acc);
    }
#pragma unroll
    for (int i = 0; i < 8; ++i) {
      int l = lt * 16 + i + 8 * hi;
      int s = st * 16 + col;
      float v = (s <= l) ? acc[i] * expf(sAcum[l] - sAcum[s]) : 0.f;
      sM[l * 72 + s] = (__bf16)v;
    }
  }
  __syncthreads();
  {  // async-stage incoming state ([p][n] in global) into sBS as [p][n]
    size_t base = ((size_t)(b * NCH + c) * NH + h) * (NSTATE * HP);
    for (int i = tid; i < HP * (NSTATE / 8); i += 256) {
      int p = i >> 4, ng = (i & 15) * 8;
      async_g2l_b128(lds_off(sBS + p * 136 + ng),
                     sInG + base + (size_t)p * NSTATE + ng);
    }
    wait_async0();
  }
  __syncthreads();
  const float dcoef = Dp[h];
#pragma unroll
  for (int q = 0; q < 2; ++q) {
    int ti = w * 2 + q, lt = ti >> 2, pt = ti & 3;
    v8f accD = {}, accO = {};
#pragma unroll
    for (int s0 = 0; s0 < QCH; s0 += 32) {
      v16bf a  = load_frag_a(sM, 72, lt * 16, s0, lane);
      v16bf bb = load_frag_bt(sH, 72, s0, pt * 16, lane);
      accD = wmma_bf16(a, bb, accD);
    }
#pragma unroll
    for (int n0 = 0; n0 < NSTATE; n0 += 32) {
      v16bf a  = load_frag_a(sC, 136, lt * 16, n0, lane);
      v16bf bb = load_frag_bt(sBS, 136, n0, pt * 16, lane);
      accO = wmma_bf16(a, bb, accO);
    }
#pragma unroll
    for (int i = 0; i < 8; ++i) {
      int l = lt * 16 + i + 8 * hi;
      int p = pt * 16 + col;
      size_t t = (size_t)(b * S_LEN + t0 + l);
      float h0 = hBC[t * CONV_DIM + h * HP + p];
      yb[(t * NH + h) * HP + p] = accD[i] + expf(sAcum[l]) * accO[i] + dcoef * h0;
    }
  }
}

// ----------------------------------------------------------------------------
// gating + RMSNorm -> bf16 activations
// ----------------------------------------------------------------------------
__global__ void __launch_bounds__(256)
gate_rms_kernel(const float* __restrict__ yb, const float* __restrict__ proj,
                const float* __restrict__ nw, __bf16* __restrict__ gb) {
  __shared__ float red[256];
  const int row = blockIdx.x, tid = threadIdx.x;
  float v[8], local = 0.f;
#pragma unroll
  for (int j = 0; j < 8; ++j) {
    int i = tid + j * 256;
    float g   = proj[(size_t)row * PROJ + i];
    float val = yb[(size_t)row * INTER + i] * (g / (1.f + expf(-g)));
    v[j] = val;
    local += val * val;
  }
  red[tid] = local;
  __syncthreads();
  for (int off = 128; off > 0; off >>= 1) {
    if (tid < off) red[tid] += red[tid + off];
    __syncthreads();
  }
  float rms = rsqrtf(red[0] / (float)INTER + 1e-5f);
#pragma unroll
  for (int j = 0; j < 8; ++j) {
    int i = tid + j * 256;
    gb[(size_t)row * INTER + i] = (__bf16)(v[j] * rms * nw[i]);
  }
}

extern "C" void kernel_launch(void* const* d_in, const int* in_sizes, int n_in,
                              void* d_out, int out_size, void* d_ws, size_t ws_size,
                              hipStream_t stream) {
  (void)in_sizes; (void)n_in; (void)out_size; (void)ws_size;
  const float* x       = (const float*)d_in[0];
  const float* W_in    = (const float*)d_in[1];
  const float* b_in    = (const float*)d_in[2];
  const float* conv_w  = (const float*)d_in[3];
  const float* dt_bias = (const float*)d_in[4];
  const float* A_log   = (const float*)d_in[5];
  const float* Dp      = (const float*)d_in[6];
  const float* norm_w  = (const float*)d_in[7];
  const float* W_out   = (const float*)d_in[8];
  const float* b_out   = (const float*)d_in[9];

  char* ws = (char*)d_ws;
  size_t off = 0;
  auto alloc = [&](size_t bytes) { char* p = ws + off; off += (bytes + 255) & ~(size_t)255; return p; };
  float*  proj  = (float*)alloc((size_t)ROWS * PROJ * 4);
  float*  hBC   = (float*)alloc((size_t)ROWS * CONV_DIM * 4);
  float*  dts   = (float*)alloc((size_t)ROWS * NH * 4);
  float*  Alast = (float*)alloc((size_t)B_SZ * NCH * NH * 4);
  __bf16* locS  = (__bf16*)alloc((size_t)B_SZ * NCH * NH * NSTATE * HP * 2);
  __bf16* sIn   = (__bf16*)alloc((size_t)B_SZ * NCH * NH * NSTATE * HP * 2);
  float*  yb    = (float*)alloc((size_t)ROWS * INTER * 4);
  __bf16* gb    = (__bf16*)alloc((size_t)ROWS * INTER * 2);
  __bf16* xbf   = (__bf16*)alloc((size_t)ROWS * HID * 2);
  __bf16* Wibf  = (__bf16*)alloc((size_t)PROJ_PAD * HID * 2);
  __bf16* Wobf  = (__bf16*)alloc((size_t)HID * INTER * 2);

  {
    long n;
    n = (long)ROWS * HID;
    f32_to_bf16_kernel<<<(unsigned)((n + 255) / 256), 256, 0, stream>>>(x, xbf, n, n);
    long nsrc = (long)PROJ * HID, ntot = (long)PROJ_PAD * HID;
    f32_to_bf16_kernel<<<(unsigned)((ntot + 255) / 256), 256, 0, stream>>>(W_in, Wibf, nsrc, ntot);
    n = (long)HID * INTER;
    f32_to_bf16_kernel<<<(unsigned)((n + 255) / 256), 256, 0, stream>>>(W_out, Wobf, n, n);
  }

  gemm_bf16<<<dim3(ROWS / 64, PROJ_PAD / 64), 256, 0, stream>>>(
      xbf, Wibf, b_in, proj, ROWS, PROJ, HID);
  conv_silu_kernel<<<(unsigned)(((size_t)ROWS * CONV_DIM + 255) / 256), 256, 0, stream>>>(
      proj, conv_w, hBC);
  dt_softplus_kernel<<<(ROWS * NH + 255) / 256, 256, 0, stream>>>(proj, dt_bias, dts);
  ssd_pass1<<<B_SZ * NCH * NH, 256, 0, stream>>>(hBC, dts, A_log, locS, Alast);
  ssd_pass2<<<B_SZ * NH, 256, 0, stream>>>(locS, Alast, sIn);
  ssd_pass3<<<B_SZ * NCH * NH, 256, 0, stream>>>(hBC, dts, A_log, sIn, Dp, yb);
  gate_rms_kernel<<<ROWS, 256, 0, stream>>>(yb, proj, norm_w, gb);
  gemm_bf16<<<dim3(ROWS / 64, HID / 64), 256, 0, stream>>>(
      gb, Wobf, b_out, (float*)d_out, ROWS, HID, INTER);
}
